// SelfAttention_65317862637785
// MI455X (gfx1250) — matrix operand order repaired and checked by reading, not compiled, not run on hardware
//
#include <hip/hip_runtime.h>

#define BB 8
#define CH 256
#define NN 4096
#define JT 32

typedef __attribute__((ext_vector_type(16))) _Float16     v16h;
typedef __attribute__((ext_vector_type(8)))  _Float16     v8h;
typedef __attribute__((ext_vector_type(8)))  float        v8f;
typedef __attribute__((ext_vector_type(4)))  unsigned int v4u;
typedef __attribute__((ext_vector_type(8)))  int          v8i;
typedef __attribute__((ext_vector_type(4)))  int          v4i;

#if defined(__HIP_DEVICE_COMPILE__) && __has_builtin(__builtin_amdgcn_tensor_load_to_lds)
#define HAVE_TDM 1
#else
#define HAVE_TDM 0
#endif

#define WMMA_F32_F16(a,b,c) \
  __builtin_amdgcn_wmma_f32_16x16x32_f16(false,(a),false,(b),(short)0,(c),false,false)

static __device__ __forceinline__ v16h cat8(v8h lo, v8h hi) {
  return __builtin_shufflevector(lo, hi, 0,1,2,3,4,5,6,7,8,9,10,11,12,13,14,15);
}

// A fragment (16x32 f16), element (m,k) at base[m*stride + k0 + k]
static __device__ __forceinline__ v16h load_frag_a(const _Float16* base, int stride,
                                                   int k0, int lane) {
  int row  = lane & 15;
  int koff = (lane & 16) ? 8 : 0;
  const _Float16* p = base + row * stride + k0 + koff;
  v8h lo = *(const v8h*)(p);
  v8h hi = *(const v8h*)(p + 16);
  return cat8(lo, hi);
}

// B fragment (32x16 f16), element (k,n) at base[n*stride + k0 + k]
static __device__ __forceinline__ v16h load_frag_b(const _Float16* base, int stride,
                                                   int k0, int lane) {
  int col = lane & 15;
  int kb  = (lane & 16) ? 8 : 0;
  const _Float16* p = base + col * stride + k0 + kb;
  v8h lo = *(const v8h*)(p);
  v8h hi = *(const v8h*)(p + 16);
  return cat8(lo, hi);
}

#if HAVE_TDM
// Issue a 2D tile DMA: global (nRows x rowBytes, row stride strideBytes) -> LDS contiguous.
// D# layout per CDNA5 ISA: type=2, data_size=1B,
// tensor dims == tile dims (no OOB), workgroup_mask=0 (not in a cluster).
// clang-23 form: 6 args (v4u, v8i, v4i, v4i, v8i, i32 cpol).
static __device__ __forceinline__ void tdm_load_2d(unsigned lds_addr, const void* gptr,
                                                   unsigned rowBytes, unsigned nRows,
                                                   unsigned strideBytes) {
  unsigned long long ga = (unsigned long long)(size_t)gptr;
  v4u g0;
  g0[0] = 1u;                                               // count=1, user descriptor
  g0[1] = lds_addr;                                         // lds_addr [63:32]
  g0[2] = (unsigned)(ga & 0xFFFFFFFFu);                     // global_addr lo
  g0[3] = (unsigned)((ga >> 32) & 0x01FFFFFFu) | (2u << 30); // global_addr hi | type=2
  v8i g1;
  g1[0] = 0;                                                // wg_mask=0, data_size=0(1B)
  g1[1] = (int)((rowBytes & 0xFFFFu) << 16);                // tensor_dim0 lo16
  g1[2] = (int)((rowBytes >> 16) & 0xFFFFu)                 // tensor_dim0 hi16
        | (int)((nRows & 0xFFFFu) << 16);                   // tensor_dim1 lo16
  g1[3] = (int)((nRows >> 16) & 0xFFFFu)                    // tensor_dim1 hi16
        | (int)((rowBytes & 0xFFFFu) << 16);                // tile_dim0
  g1[4] = (int)(nRows & 0xFFFFu);                           // tile_dim1 (tile_dim2=0)
  g1[5] = (int)strideBytes;                                 // tensor_dim0_stride lo32
  g1[6] = 0;
  g1[7] = 0;
  v4i gz  = {0, 0, 0, 0};
  v8i gz8 = {0, 0, 0, 0, 0, 0, 0, 0};
  __builtin_amdgcn_tensor_load_to_lds(g0, g1, gz, gz, gz8, 0);
}
#endif

// ---------------- kernel 1: convert weights to f16 ----------------
__global__ void wcvt_kernel(const float* __restrict__ wq, const float* __restrict__ wk,
                            const float* __restrict__ wv, _Float16* __restrict__ wh) {
  int i = blockIdx.x * blockDim.x + threadIdx.x;
  if (i >= 3 * CH * CH) return;
  int m = i / (CH * CH), r = i % (CH * CH);
  const float* src = (m == 0) ? wq : (m == 1) ? wk : wv;
  wh[i] = (_Float16)src[r];
}

// ---------------- kernel 2: fused QKV projection (WMMA) ----------------
__global__ void __launch_bounds__(128) qkv_kernel(
    const float* __restrict__ x, const _Float16* __restrict__ wh,
    const float* __restrict__ bq, const float* __restrict__ bk, const float* __restrict__ bv,
    _Float16* __restrict__ qt, _Float16* __restrict__ kt, _Float16* __restrict__ vc)
{
  __shared__ __align__(32) _Float16 xs[64 * CH];   // [n][c], 32KB
  int b   = blockIdx.y;
  int n0  = blockIdx.x * 64;
  int tid = threadIdx.x;

  for (int idx = tid; idx < CH * 64; idx += 128) {
    int c = idx >> 6, n = idx & 63;
    xs[n * CH + c] = (_Float16)x[((size_t)b * CH + c) * NN + n0 + n];
  }
  __syncthreads();

  int lane = tid & 31, wave = tid >> 5;
  int hig  = (lane >> 4) & 1;

  for (int m = 0; m < 3; ++m) {
    const _Float16* wm = wh + m * CH * CH;
    const float* bias  = (m == 0) ? bq : (m == 1) ? bk : bv;
    _Float16* outp     = (m == 0) ? qt : (m == 1) ? kt : vc;
    for (int os = 0; os < 4; ++os) {
      int o0 = wave * 64 + os * 16;
      v16h aw[8];
      #pragma unroll
      for (int cc = 0; cc < 8; ++cc)
        aw[cc] = load_frag_a(wm + o0 * CH, CH, cc * 32, lane);
      float bb[8];
      #pragma unroll
      for (int r = 0; r < 8; ++r) bb[r] = bias[o0 + r + 8 * hig];
      for (int ns = 0; ns < 4; ++ns) {
        v8f acc0 = {}, acc1 = {};                 // two independent WMMA chains
        #pragma unroll
        for (int cc = 0; cc < 8; cc += 2) {
          v16h bx0 = load_frag_b(xs + (ns * 16) * CH, CH, cc * 32, lane);
          v16h bx1 = load_frag_b(xs + (ns * 16) * CH, CH, (cc + 1) * 32, lane);
          acc0 = WMMA_F32_F16(aw[cc],     bx0, acc0);
          acc1 = WMMA_F32_F16(aw[cc + 1], bx1, acc1);
        }
        int n = n0 + ns * 16 + (lane & 15);
        if (m < 2) {                              // [b][n][c]: pack 2 halves -> b32
          _Float16* rowp = outp + ((size_t)b * NN + n) * CH + o0 + 8 * hig;
          #pragma unroll
          for (int t = 0; t < 4; ++t) {
            union { _Float16 h[2]; unsigned u; } pk;
            pk.h[0] = (_Float16)(acc0[2 * t]     + acc1[2 * t]     + bb[2 * t]);
            pk.h[1] = (_Float16)(acc0[2 * t + 1] + acc1[2 * t + 1] + bb[2 * t + 1]);
            *(unsigned*)(rowp + 2 * t) = pk.u;
          }
        } else {                                  // [b][c][n]
          #pragma unroll
          for (int r = 0; r < 8; ++r) {
            int o = o0 + r + 8 * hig;
            outp[((size_t)b * CH + o) * NN + n] =
                (_Float16)(acc0[r] + acc1[r] + bb[r]);
          }
        }
      }
    }
  }
}

// ---------------- kernel 3: flash attention (online softmax, WMMA, TDM) ----------------
#if HAVE_TDM
#define NBUF 2
#else
#define NBUF 1
#endif

__global__ void __launch_bounds__(128) attn_kernel(
    const _Float16* __restrict__ qt, const _Float16* __restrict__ kt,
    const _Float16* __restrict__ vc, float* __restrict__ out)
{
  __shared__ __align__(32) _Float16 ks[NBUF][JT * CH];     // [j][c]
  __shared__ __align__(32) _Float16 vs[NBUF][CH * JT];     // [c][j]
  __shared__ __align__(32) _Float16 ps[4][16 * JT];        // per-wave P tile

  int b    = blockIdx.y;
  int tid  = threadIdx.x, lane = tid & 31, wave = tid >> 5;
  int hig  = (lane >> 4) & 1;
  int i0   = blockIdx.x * 64 + wave * 16;

  // resident Q fragments: 16 rows x 256 c
  const _Float16* qbase = qt + ((size_t)b * NN + i0) * CH;
  v16h qa[8];
  #pragma unroll
  for (int cc = 0; cc < 8; ++cc) qa[cc] = load_frag_a(qbase, CH, cc * 32, lane);

  v8f zero = {};
  v8f oacc[16];
  #pragma unroll
  for (int t = 0; t < 16; ++t) oacc[t] = zero;
  float mrow[8], lrow[8];
  #pragma unroll
  for (int r = 0; r < 8; ++r) { mrow[r] = -3.0e38f; lrow[r] = 0.0f; }

  const _Float16* kglob = kt + (size_t)b * NN * CH;
  const _Float16* vglob = vc + (size_t)b * CH * NN;

#if HAVE_TDM
  int buf = 0;
  // prologue: DMA tile 0 into buffer 0 (K by wave0, V by wave1); TDM ignores EXEC
  if (tid == 0)
    tdm_load_2d((unsigned)(size_t)&ks[0][0], kglob, CH * 2, JT, CH * 2);
  else if (tid == 32)
    tdm_load_2d((unsigned)(size_t)&vs[0][0], vglob, JT * 2, CH, NN * 2);
  __builtin_amdgcn_s_wait_tensorcnt(0);
  __syncthreads();
#endif

  for (int j0 = 0; j0 < NN; j0 += JT) {
#if HAVE_TDM
    int nbuf = buf ^ 1;
    if (j0 + JT < NN) {                    // overlap next tile's DMA with compute
      if (tid == 0)
        tdm_load_2d((unsigned)(size_t)&ks[nbuf][0],
                    kglob + (size_t)(j0 + JT) * CH, CH * 2, JT, CH * 2);
      else if (tid == 32)
        tdm_load_2d((unsigned)(size_t)&vs[nbuf][0],
                    vglob + (size_t)(j0 + JT), JT * 2, CH, NN * 2);
    }
    const _Float16* kst = ks[buf];
    const _Float16* vst = vs[buf];
#else
    {  // synchronous staging fallback
      const v8h* ksrc = (const v8h*)(kglob + (size_t)j0 * CH);
      v8h* kdst = (v8h*)ks[0];
      for (int idx = tid; idx < JT * CH / 8; idx += 128) kdst[idx] = ksrc[idx];
      for (int idx = tid; idx < CH * (JT / 8); idx += 128) {
        int c = idx >> 2, seg = idx & 3;
        ((v8h*)vs[0])[idx] = *(const v8h*)(vglob + (size_t)c * NN + j0 + seg * 8);
      }
      if (j0 + JT < NN)
        __builtin_prefetch(kglob + (size_t)(j0 + JT) * CH + tid * 64, 0, 0);
      __syncthreads();
    }
    const _Float16* kst = ks[0];
    const _Float16* vst = vs[0];
#endif

    // S = Q^T K : two 16x16 tiles, K-depth 256, two independent WMMA chains
    v8f s0 = zero, s1 = zero;
    #pragma unroll
    for (int cc = 0; cc < 8; ++cc) {
      v16h kb0 = load_frag_b(kst,           CH, cc * 32, lane);
      v16h kb1 = load_frag_b(kst + 16 * CH, CH, cc * 32, lane);
      s0 = WMMA_F32_F16(qa[cc], kb0, s0);
      s1 = WMMA_F32_F16(qa[cc], kb1, s1);
    }

    // online softmax; D-layout rows M = r + 8*hig, columns = lane&15
    float scale[8];
    _Float16* psw = ps[wave];
    #pragma unroll
    for (int r = 0; r < 8; ++r) {
      float mx = fmaxf(s0[r], s1[r]);
      #pragma unroll
      for (int d = 8; d >= 1; d >>= 1) mx = fmaxf(mx, __shfl_xor(mx, d, 32));
      float mn   = fmaxf(mrow[r], mx);
      float corr = __expf(mrow[r] - mn);
      float p0   = __expf(s0[r] - mn);
      float p1   = __expf(s1[r] - mn);
      float rs   = p0 + p1;
      #pragma unroll
      for (int d = 8; d >= 1; d >>= 1) rs += __shfl_xor(rs, d, 32);
      lrow[r]  = lrow[r] * corr + rs;
      mrow[r]  = mn;
      scale[r] = corr;
      int irow = r + 8 * hig;
      int jc   = lane & 15;
      psw[irow * JT + jc]      = (_Float16)p0;   // re-layout D -> A via LDS
      psw[irow * JT + jc + 16] = (_Float16)p1;
    }
    #pragma unroll
    for (int t = 0; t < 16; ++t)
      #pragma unroll
      for (int r = 0; r < 8; ++r) oacc[t][r] *= scale[r];

    asm volatile("s_wait_dscnt 0" ::: "memory");

    // O += P * V^T : 16 independent accumulator tiles
    v16h pa = load_frag_a(psw, JT, 0, lane);
    #pragma unroll
    for (int cs = 0; cs < 16; ++cs) {
      v16h bvf = load_frag_b(vst + (cs * 16) * JT, JT, 0, lane);
      oacc[cs] = WMMA_F32_F16(pa, bvf, oacc[cs]);
    }

#if HAVE_TDM
    __builtin_amdgcn_s_wait_tensorcnt(0);   // collect next tile's DMA
    __syncthreads();
    buf = nbuf;
#else
    __syncthreads();
#endif
  }

  // epilogue: out[b][c][i] = O / l ; consecutive i across regs -> b64 stores
  float rinv[8];
  #pragma unroll
  for (int r = 0; r < 8; ++r) rinv[r] = 1.0f / lrow[r];
  #pragma unroll
  for (int cs = 0; cs < 16; ++cs) {
    int c = cs * 16 + (lane & 15);
    float* rowp = out + ((size_t)b * CH + c) * NN + i0 + 8 * hig;
    #pragma unroll
    for (int t = 0; t < 4; ++t) {
      float2 v;
      v.x = oacc[cs][2 * t]     * rinv[2 * t];
      v.y = oacc[cs][2 * t + 1] * rinv[2 * t + 1];
      *(float2*)(rowp + 2 * t) = v;
    }
  }
}

extern "C" void kernel_launch(void* const* d_in, const int* in_sizes, int n_in,
                              void* d_out, int out_size, void* d_ws, size_t ws_size,
                              hipStream_t stream) {
  const float* x  = (const float*)d_in[0];
  const float* wq = (const float*)d_in[1];
  const float* bq = (const float*)d_in[2];
  const float* wk = (const float*)d_in[3];
  const float* bk = (const float*)d_in[4];
  const float* wv = (const float*)d_in[5];
  const float* bv = (const float*)d_in[6];
  float* out = (float*)d_out;

  // workspace (f16): W (3*C*C) | Qt [b][n][c] | Kt [b][n][c] | Vc [b][c][n]
  _Float16* wh = (_Float16*)d_ws;
  _Float16* qt = wh + (size_t)3 * CH * CH;
  _Float16* kt = qt + (size_t)BB * NN * CH;
  _Float16* vc = kt + (size_t)BB * NN * CH;

  wcvt_kernel<<<(3 * CH * CH + 255) / 256, 256, 0, stream>>>(wq, wk, wv, wh);
  qkv_kernel<<<dim3(NN / 64, BB), 128, 0, stream>>>(x, wh, bq, bk, bv, qt, kt, vc);
  attn_kernel<<<dim3(NN / 64, BB), 128, 0, stream>>>(qt, kt, vc, out);
}
